// Attention_24550033064049
// MI455X (gfx1250) — matrix operand order repaired and checked by reading, not compiled, not run on hardware
//
#include <hip/hip_runtime.h>
#include <hip/hip_bf16.h>
#include <math.h>

typedef __attribute__((ext_vector_type(16))) _Float16 v16h;
typedef __attribute__((ext_vector_type(8)))  _Float16 v8h;
typedef __attribute__((ext_vector_type(4)))  _Float16 v4h;
typedef __attribute__((ext_vector_type(2)))  _Float16 v2h;
typedef __attribute__((ext_vector_type(8)))  float    v8f;

#define B_    4
#define H_    16
#define S_    2048
#define D_    128
#define BR    128     // Q rows per workgroup (16 per wave x 8 waves)
#define BC    32      // KV rows per tile
#define WAVES 8
#define TPB   256
#define KPITCH 136    // halfs: 128 + 8 pad (272B rows, 8B-aligned for b64 stores)
#define VPITCH 40     // halfs: 32 + 8 pad for transposed V
#define PPITCH 40     // halfs: 32 + 8 pad for per-wave P scratch

// packed f32->f16 convert with the result re-typed to our _Float16 vectors
__device__ __forceinline__ v2h cvt_pk_h2(float a, float b) {
    return __builtin_bit_cast(v2h, __builtin_amdgcn_cvt_pkrtz(a, b));
}

// ---- 16-lane row reductions via DPP16 (wave32: DPP row == our 16-lane group).
// quad_perm {1,0,3,2}=0xB1 (xor1), {2,3,0,1}=0x4E (xor2), then row_ror:4/8
// (valid for commutative reduce once quads are uniform).
__device__ __forceinline__ float rowRedMax16(float x) {
    float y;
    y = __int_as_float(__builtin_amdgcn_update_dpp(0, __float_as_int(x), 0xB1, 0xF, 0xF, true));
    x = fmaxf(x, y);
    y = __int_as_float(__builtin_amdgcn_update_dpp(0, __float_as_int(x), 0x4E, 0xF, 0xF, true));
    x = fmaxf(x, y);
    y = __int_as_float(__builtin_amdgcn_update_dpp(0, __float_as_int(x), 0x124, 0xF, 0xF, true));
    x = fmaxf(x, y);
    y = __int_as_float(__builtin_amdgcn_update_dpp(0, __float_as_int(x), 0x128, 0xF, 0xF, true));
    x = fmaxf(x, y);
    return x;
}
__device__ __forceinline__ float rowRedSum16(float x) {
    float y;
    y = __int_as_float(__builtin_amdgcn_update_dpp(0, __float_as_int(x), 0xB1, 0xF, 0xF, true));
    x += y;
    y = __int_as_float(__builtin_amdgcn_update_dpp(0, __float_as_int(x), 0x4E, 0xF, 0xF, true));
    x += y;
    y = __int_as_float(__builtin_amdgcn_update_dpp(0, __float_as_int(x), 0x124, 0xF, 0xF, true));
    x += y;
    y = __int_as_float(__builtin_amdgcn_update_dpp(0, __float_as_int(x), 0x128, 0xF, 0xF, true));
    x += y;
    return x;
}

__global__ __launch_bounds__(TPB)
void fa_fwd_kernel(const float* __restrict__ Q,
                   const float* __restrict__ K,
                   const float* __restrict__ V,
                   const int*   __restrict__ padMask,
                   float* __restrict__ O)
{
    __shared__ _Float16 ldsK [2][BC][KPITCH];     // K tile, row-major [kv][d]
    __shared__ _Float16 ldsVt[2][D_][VPITCH];     // V tile, transposed [d][kv]
    __shared__ _Float16 ldsP [WAVES][16][PPITCH]; // per-wave P tile [m][kv]
    __shared__ int      ldsValid[2][BC];          // pad-mask validity per kv col
    __shared__ int      ldsAllValid[2];           // 1 if whole tile pad-valid

    const int lane = threadIdx.x & 31;
    const int wave = threadIdx.x >> 5;
    const int n16  = lane & 15;   // column coordinate within 16-wide tiles
    const int hi   = lane >> 4;   // half-wave select

    const int qTile = blockIdx.x;
    const int h     = blockIdx.y;
    const int b     = blockIdx.z;

    const size_t headOff = ((size_t)(b * H_ + h)) * (size_t)S_ * (size_t)D_;
    const float* qh = Q + headOff;
    const float* kh = K + headOff;
    const float* vh = V + headOff;
    float*       oh = O + headOff;
    const int*   ph = padMask + (size_t)b * S_;   // pad_mask [B,1,1,S]

    const int   qRow0  = qTile * BR + wave * 16;  // first Q row owned by this wave
    // fold (1/sqrt(D)) * log2(e) into Q so softmax works in the exp2 domain
    const float qscale = 0.08838834764831845f * 1.4426950408889634f;

    // ---------------- Q: 4 A-fragments (16x32 f16), resident for whole kernel
    // A layout: lane m = lane%16; half h -> k = (h/8)*16 + hi*8 + (h%8)
    v16h aQ[4];
    {
        const float* qrow = qh + (size_t)(qRow0 + n16) * D_;
        #pragma unroll
        for (int kc = 0; kc < 4; ++kc) {
            #pragma unroll
            for (int g = 0; g < 2; ++g) {
                const float* src = qrow + kc * 32 + g * 16 + hi * 8;
                float4 x0 = *(const float4*)(src);
                float4 x1 = *(const float4*)(src + 4);
                aQ[kc][g * 8 + 0] = (_Float16)(x0.x * qscale);
                aQ[kc][g * 8 + 1] = (_Float16)(x0.y * qscale);
                aQ[kc][g * 8 + 2] = (_Float16)(x0.z * qscale);
                aQ[kc][g * 8 + 3] = (_Float16)(x0.w * qscale);
                aQ[kc][g * 8 + 4] = (_Float16)(x1.x * qscale);
                aQ[kc][g * 8 + 5] = (_Float16)(x1.y * qscale);
                aQ[kc][g * 8 + 6] = (_Float16)(x1.z * qscale);
                aQ[kc][g * 8 + 7] = (_Float16)(x1.w * qscale);
            }
        }
    }

    // ---------------- running softmax state (log2 domain) + O accumulators
    // C-layout broadcast convention: index r -> row r (lanes 0-15), r+8 (16-31)
    float mrun[8], lrun[8];
    v8f   oAcc[8];
    #pragma unroll
    for (int r = 0; r < 8; ++r) { mrun[r] = -INFINITY; lrun[r] = 0.0f; }
    #pragma unroll
    for (int df = 0; df < 8; ++df)
        #pragma unroll
        for (int r = 0; r < 8; ++r) oAcc[df][r] = 0.0f;

    const int nTiles = (qTile + 1) * (BR / BC);  // causal upper bound

    // ---------------- staging helpers (double-buffered K/V in LDS)
    float4 kreg[4], vreg[4];
    int    padreg = 1;

    auto stageRegs = [&](int kvBase) {
        const float4* k4 = (const float4*)(kh + (size_t)kvBase * D_);
        const float4* v4 = (const float4*)(vh + (size_t)kvBase * D_);
        #pragma unroll
        for (int it = 0; it < 4; ++it) {
            const int e = threadIdx.x + it * TPB;
            kreg[it] = k4[e];
            vreg[it] = v4[e];
        }
        padreg = (threadIdx.x < BC) ? ph[kvBase + threadIdx.x] : 1;
    };
    auto commitLds = [&](int buf) {
        #pragma unroll
        for (int it = 0; it < 4; ++it) {
            const int e = threadIdx.x + it * TPB;
            const int r = (e * 4) / D_;
            const int c = (e * 4) % D_;
            // packed f32->f16 converts (v_cvt_pk_f16_f32)
            v2h ka = cvt_pk_h2(kreg[it].x, kreg[it].y);
            v2h kb = cvt_pk_h2(kreg[it].z, kreg[it].w);
            v4h kw;
            kw[0] = ka[0]; kw[1] = ka[1]; kw[2] = kb[0]; kw[3] = kb[1];
            *(v4h*)&ldsK[buf][r][c] = kw;   // single 8B ds_store
            v2h va = cvt_pk_h2(vreg[it].x, vreg[it].y);
            v2h vb = cvt_pk_h2(vreg[it].z, vreg[it].w);
            ldsVt[buf][c + 0][r] = va[0];
            ldsVt[buf][c + 1][r] = va[1];
            ldsVt[buf][c + 2][r] = vb[0];
            ldsVt[buf][c + 3][r] = vb[1];
        }
        if (threadIdx.x < BC) ldsValid[buf][threadIdx.x] = padreg;
        if (wave == 0) {
            // wave 0 lanes hold the BC pad flags -> single all-valid bit
            unsigned long long bal = __ballot(padreg != 0);
            if (lane == 0)
                ldsAllValid[buf] = ((bal & 0xFFFFFFFFull) == 0xFFFFFFFFull) ? 1 : 0;
        }
    };

    // prologue: fill buffer 0
    stageRegs(0);
    commitLds(0);

    for (int j = 0; j < nTiles; ++j) {
        const int  cur      = j & 1;
        const int  kvBase   = j * BC;
        const bool haveNext = (j + 1) < nTiles;

        __syncthreads();   // buf[cur] visible; buf[cur^1] fully consumed

        // issue next tile's global loads now; compute below hides the latency
        if (haveNext) stageRegs(kvBase + BC);

        // ---- S = (Q*qscale) K^T : two 16x16 f32 tiles, K-dim 128 = 4 x 32
        v8f sAcc[2];
        #pragma unroll
        for (int nt = 0; nt < 2; ++nt) {
            v8f acc;
            #pragma unroll
            for (int i = 0; i < 8; ++i) acc[i] = 0.0f;
            #pragma unroll
            for (int kc = 0; kc < 4; ++kc) {
                const _Float16* p = &ldsK[cur][nt * 16 + n16][kc * 32 + hi * 8];
                v8h lo  = *(const v8h*)(p);
                v8h hiv = *(const v8h*)(p + 16);
                v16h bK;
                #pragma unroll
                for (int i = 0; i < 8; ++i) { bK[i] = lo[i]; bK[i + 8] = hiv[i]; }
                acc = __builtin_amdgcn_wmma_f32_16x16x32_f16(
                        false, aQ[kc], false, bK, (short)0, acc, false, false);
            }
            sAcc[nt] = acc;
        }

        // ---- masking: force a truly scalar (wave-uniform) fast-path branch.
        // Tile is fully causal-legal iff last kv col <= first q row of this
        // wave; the predicate is wave-uniform, readfirstlane makes it SGPR so
        // the compiler emits s_cbranch instead of an EXEC-mask dance.
        const int fullCausal = ((kvBase + BC - 1) <= qRow0) ? 1 : 0;
        const int needMask   = __builtin_amdgcn_readfirstlane(
                                   (!fullCausal) | (ldsAllValid[cur] == 0));
        if (needMask) {
            const int valid0 = ldsValid[cur][n16];
            const int valid1 = ldsValid[cur][16 + n16];
            #pragma unroll
            for (int nt = 0; nt < 2; ++nt) {
                const int  cg  = kvBase + nt * 16 + n16;
                const bool cOK = (nt ? valid1 : valid0) != 0;
                #pragma unroll
                for (int r = 0; r < 8; ++r) {
                    const int rg = qRow0 + r + 8 * hi;
                    sAcc[nt][r] = (cOK && (cg <= rg)) ? sAcc[nt][r] : -INFINITY;
                }
            }
        }

        // ---- online softmax: row max via DPP16, exp2 for weights
        float mNew[8], mSub[8], alpha[8];
        #pragma unroll
        for (int r = 0; r < 8; ++r) {
            float t = rowRedMax16(fmaxf(sAcc[0][r], sAcc[1][r]));
            mNew[r]  = fmaxf(mrun[r], t);
            mSub[r]  = fmaxf(mNew[r], -3.0e38f);   // avoid inf-inf NaN on dead rows
            alpha[r] = __builtin_amdgcn_exp2f(mrun[r] - mSub[r]);
            mrun[r]  = mNew[r];
        }

        #pragma unroll
        for (int nt = 0; nt < 2; ++nt)
            #pragma unroll
            for (int r = 0; r < 8; ++r)
                sAcc[nt][r] = __builtin_amdgcn_exp2f(sAcc[nt][r] - mSub[r]);

        #pragma unroll
        for (int r = 0; r < 8; ++r) {
            const float s = rowRedSum16(sAcc[0][r] + sAcc[1][r]);
            lrun[r] = lrun[r] * alpha[r] + s;
        }
        #pragma unroll
        for (int df = 0; df < 8; ++df)
            #pragma unroll
            for (int r = 0; r < 8; ++r)
                oAcc[df][r] *= alpha[r];

        // ---- relayout P: C-layout f32 -> A-layout f16 via wave-private LDS
        #pragma unroll
        for (int nt = 0; nt < 2; ++nt)
            #pragma unroll
            for (int r = 0; r < 8; ++r)
                ldsP[wave][r + 8 * hi][nt * 16 + n16] = (_Float16)sAcc[nt][r];

        asm volatile("s_wait_dscnt 0x0" ::: "memory"); // wave-local store->load

        v16h pFrag;
        {
            const _Float16* pp = &ldsP[wave][n16][0];
            #pragma unroll
            for (int g = 0; g < 2; ++g) {
                v8h c = *(const v8h*)(pp + g * 16 + hi * 8);
                #pragma unroll
                for (int i = 0; i < 8; ++i) pFrag[g * 8 + i] = c[i];
            }
        }

        // ---- O += P V : 8 d-tiles of 16 cols, inner K = 32 (one WMMA each)
        #pragma unroll
        for (int df = 0; df < 8; ++df) {
            const _Float16* vp = &ldsVt[cur][df * 16 + n16][hi * 8];
            v8h lo  = *(const v8h*)(vp);
            v8h hiv = *(const v8h*)(vp + 16);
            v16h bV;
            #pragma unroll
            for (int i = 0; i < 8; ++i) { bV[i] = lo[i]; bV[i + 8] = hiv[i]; }
            oAcc[df] = __builtin_amdgcn_wmma_f32_16x16x32_f16(
                          false, pFrag, false, bV, (short)0, oAcc[df], false, false);
        }

        // commit next tile into the other buffer (separated from prior readers
        // of buf[cur^1] by the barrier at the top of this iteration)
        if (haveNext) commitLds(cur ^ 1);
    }

    // ---------------- epilogue: O / l, store f32
    float inv[8];
    #pragma unroll
    for (int r = 0; r < 8; ++r)
        inv[r] = (lrun[r] > 0.0f) ? (1.0f / lrun[r]) : 0.0f;

    #pragma unroll
    for (int df = 0; df < 8; ++df) {
        #pragma unroll
        for (int r = 0; r < 8; ++r) {
            const int m = r + 8 * hi;
            const int d = df * 16 + n16;
            oh[(size_t)(qRow0 + m) * D_ + d] = oAcc[df][r] * inv[r];
        }
    }
}

extern "C" void kernel_launch(void* const* d_in, const int* in_sizes, int n_in,
                              void* d_out, int out_size, void* d_ws, size_t ws_size,
                              hipStream_t stream)
{
    const float* q   = (const float*)d_in[0];
    const float* k   = (const float*)d_in[1];
    const float* v   = (const float*)d_in[2];
    // d_in[3] = attn_mask (lower-triangular) -- computed analytically in-kernel
    const int*   pad = (const int*)d_in[4];
    float*       out = (float*)d_out;

    dim3 grid(S_ / BR, H_, B_);
    dim3 block(TPB);
    hipLaunchKernelGGL(fa_fwd_kernel, grid, block, 0, stream, q, k, v, pad, out);
}
